// LinPolyLeakage_18253611008466
// MI455X (gfx1250) — matrix operand order repaired and checked by reading, not compiled, not run on hardware
//
#include <hip/hip_runtime.h>

// ---------------- problem constants ----------------
#define B_   128
#define N_   16384
#define C_   8
#define W_   10
#define L_   (N_ - W_ + 1)     // 16375
#define EPS_ 1e-5f

// FIR-as-GEMM tiling
#define KC_   32               // k-chunk blocks along N
#define NSEG_ (N_ / KC_)       // 512 n positions per block
#define NSUB_ 64               // n positions per LDS tile
#define SUBS_ (NSEG_ / NSUB_)  // 8 LDS tiles per block

typedef __attribute__((ext_vector_type(16))) __bf16          v16bf;
typedef __attribute__((ext_vector_type(8)))  float           v8f;
typedef __attribute__((ext_vector_type(8)))  unsigned short  v8u;

__device__ __forceinline__ unsigned short f2bf(float f) {
  // round-to-nearest-even f32 -> bf16
  unsigned int u = __float_as_uint(f);
  unsigned int rb = ((u >> 16) & 1u) + 0x7FFFu;
  return (unsigned short)((u + rb) >> 16);
}

__device__ __forceinline__ unsigned int pk2(float a, float b) {
  return (unsigned int)f2bf(a) | ((unsigned int)f2bf(b) << 16);
}

__device__ __forceinline__ v16bf make_frag(const unsigned short* p0,
                                           const unsigned short* p1) {
  union { v8u h[2]; v16bf bf; } u;
  u.h[0] = *(const v8u*)p0;
  u.h[1] = *(const v8u*)p1;
  return u.bf;
}

// ---------------------------------------------------------------------------
// Kernel 0: build shifted weight matrices directly in WMMA B-fragment layout.
// Bfrag[frag][lane][t], frag = (p*8 + c)*512 + k32; per lane: 16 consecutive
// K values (lane<16 -> K 0..15 of column w=lane; lane>=16 -> K 16..31 of
// column w=lane-16).  S[k, w] = w_in[c, k - w] (zero outside [0, L)).
// ---------------------------------------------------------------------------
__global__ __launch_bounds__(256) void prep_bfrag(
    const float* __restrict__ w_in_real,
    const float* __restrict__ w_in_imag,
    unsigned short* __restrict__ Bfrag) {
  int tid  = blockIdx.x * 256 + threadIdx.x;   // 0 .. 262143
  int lane = tid & 31;
  int frag = tid >> 5;                         // 0 .. 8191
  int p    = frag >> 12;                       // frags per p = 8*512 = 4096
  int rem  = frag & 4095;
  int c    = rem >> 9;
  int k32  = rem & 511;
  const float* w = p ? w_in_imag : w_in_real;
  int wcol = lane & 15;
  int kb   = (lane >> 4) * 16;
  const bool live = (wcol < W_);
  const int  jbase = k32 * 32 + kb - wcol;
  const float* wp = w + c * L_ + jbase;        // single base, imm offsets below
  unsigned short* dst = Bfrag + (size_t)frag * 512 + lane * 16;
#pragma unroll
  for (int t = 0; t < 16; ++t) {
    int j = jbase + t;
    float v = (live && j >= 0 && j < L_) ? wp[t] : 0.0f;
    dst[t] = f2bf(v);
  }
}

// ---------------------------------------------------------------------------
// Kernel 1: FIR as bf16 WMMA GEMM.
// Grid (8 b-tiles, 32 k-chunks), 256 threads = 8 waves.
// Each block stages 16b x 64n x 16cp fp32->bf16 into LDS; wave w owns
// (c,p) planes {w, w+8}; accumulates v8f C tiles and writes k-partials.
// ---------------------------------------------------------------------------
__global__ __launch_bounds__(256) void fir_wmma(
    const float* __restrict__ x,
    const unsigned short* __restrict__ Bfrag,
    float* __restrict__ Ypart) {
  __shared__ __align__(32) unsigned short Alds[16][16][NSUB_]; // [cp][bm][n]

  const int bt    = blockIdx.x;     // 0..7
  const int kc    = blockIdx.y;     // 0..31
  const int t     = threadIdx.x;
  const int lane  = t & 31;
  const int wave  = t >> 5;         // 0..7
  const int b0    = bt * 16;
  const int nbase = kc * NSEG_;

  const int cp0 = wave;
  const int cp1 = wave + 8;
  // Bfrag plane base for each owned (c,p):  frag = (p*8 + c)*512
  const size_t bplane0 = (size_t)(((cp0 & 1) * 8 + (cp0 >> 1))) * 512;
  const size_t bplane1 = (size_t)(((cp1 & 1) * 8 + (cp1 >> 1))) * 512;

  v8f acc0 = {};
  v8f acc1 = {};

  // staging decomposition: 4 cp (q) x 4 n (ng) x 4 bm (bg) per thread
  const int q  = t & 3;             // which float4 of the 16-float (c,p) row
  const int ng = (t >> 2) & 15;     // n-group of 4
  const int bg = t >> 6;            // b-group of 4

  const int half = lane >> 4;       // A: which K-run; C: which M-half
  const int mrow = lane & 15;       // A: M row; C: column (N)

  for (int s = 0; s < SUBS_; ++s) {
    const int n0 = nbase + s * NSUB_;
    // ---- stage x tile into LDS as bf16 (coalesced 16B global loads,
    //      packed 8B ds stores: 4 consecutive n per cp plane) ----
#pragma unroll
    for (int ib = 0; ib < 4; ++ib) {
      const int bm = bg * 4 + ib;
      const float* src =
          x + (((size_t)(b0 + bm) * N_ + (n0 + ng * 4)) * 16 + 4 * q);
      const float4 v0 = *(const float4*)(src);
      const float4 v1 = *(const float4*)(src + 16);
      const float4 v2 = *(const float4*)(src + 32);
      const float4 v3 = *(const float4*)(src + 48);
      uint2 pk;
      pk.x = pk2(v0.x, v1.x); pk.y = pk2(v2.x, v3.x);
      *(uint2*)&Alds[4 * q + 0][bm][ng * 4] = pk;
      pk.x = pk2(v0.y, v1.y); pk.y = pk2(v2.y, v3.y);
      *(uint2*)&Alds[4 * q + 1][bm][ng * 4] = pk;
      pk.x = pk2(v0.z, v1.z); pk.y = pk2(v2.z, v3.z);
      *(uint2*)&Alds[4 * q + 2][bm][ng * 4] = pk;
      pk.x = pk2(v0.w, v1.w); pk.y = pk2(v2.w, v3.w);
      *(uint2*)&Alds[4 * q + 3][bm][ng * 4] = pk;
    }
    if (s + 1 < SUBS_) {  // stream next tile toward L2 (global_prefetch_b8)
      __builtin_prefetch(
          x + (((size_t)(b0 + (t >> 4)) * N_ + (n0 + NSUB_ + (t & 15) * 4)) * 16),
          0, 1);
    }
    __syncthreads();

    // ---- two K=32 WMMA steps over the 64-n tile, per owned (c,p) ----
#pragma unroll
    for (int kh = 0; kh < 2; ++kh) {
      const int kloc    = kh * 32;
      const int kglob32 = ((n0 >> 5) + kh);

      const unsigned short* a0p = &Alds[cp0][mrow][kloc + half * 16];
      const unsigned short* a1p = &Alds[cp1][mrow][kloc + half * 16];
      v16bf a0 = make_frag(a0p, a0p + 8);    // K run: half*16 .. half*16+15
      v16bf a1 = make_frag(a1p, a1p + 8);

      const unsigned short* bq0 = Bfrag + (bplane0 + kglob32) * 512 + lane * 16;
      const unsigned short* bq1 = Bfrag + (bplane1 + kglob32) * 512 + lane * 16;
      v16bf bf0 = make_frag(bq0, bq0 + 8);
      v16bf bf1 = make_frag(bq1, bq1 + 8);

      acc0 = __builtin_amdgcn_wmma_f32_16x16x32_bf16(
          false, a0, false, bf0, (short)0, acc0, false, false);
      acc1 = __builtin_amdgcn_wmma_f32_16x16x32_bf16(
          false, a1, false, bf1, (short)0, acc1, false, false);
    }
    __syncthreads();
  }

  // ---- write k-partial C tiles:  Ypart[kc][cp][b(128)][w(16)] ----
  const size_t base0 = ((size_t)kc * 16 + cp0) * 128;
  const size_t base1 = ((size_t)kc * 16 + cp1) * 128;
#pragma unroll
  for (int v = 0; v < 8; ++v) {
    const int m = v + 8 * half;     // C layout: lanes>=16 hold M=8..15
    Ypart[(base0 + b0 + m) * 16 + mrow] = acc0[v];
    Ypart[(base1 + b0 + m) * 16 + mrow] = acc1[v];
  }
}

// ---------------------------------------------------------------------------
// Kernel 2: reduce k-partials -> fr/fi, polynomial leakage, coefficient
// contraction, W-reduction.  One 32-lane group per batch element.
// ---------------------------------------------------------------------------
__global__ __launch_bounds__(256) void finish(
    const float* __restrict__ Ypart,
    const float* __restrict__ w_coeff,    // (C,2,42) = (c,p,(ri,jj,d))
    const float* __restrict__ w_out_real, // (C,10)
    const float* __restrict__ w_out_imag, // (C,10)
    float* __restrict__ outbuf) {         // (B,C,2)
  __shared__ float F[8][16][W_];   // [group][cp][w]
  __shared__ float O[8][16];       // [group][cp]
  const int t = threadIdx.x, g = t >> 5, lane = t & 31;
  const int b = blockIdx.x * 8 + g;

  for (int v = lane; v < 16 * W_; v += 32) {
    const int cp = v / W_, w = v % W_;
    float s = 0.0f;
    for (int kc = 0; kc < KC_; ++kc)
      s += Ypart[(((size_t)kc * 16 + cp) * 128 + b) * 16 + w];
    F[g][cp][w] = s;
  }
  if (lane < 16) O[g][lane] = 0.0f;
  __syncthreads();

  if (lane < W_) {
    const int w = lane;
    float pr[8][3], pi[8][3];
#pragma unroll
    for (int j = 0; j < 8; ++j) {
      const float fr = F[g][2 * j + 0][w];
      const float fi = F[g][2 * j + 1][w];
      const float p1 = fr + fi;
      const float p2 = fr * fr + fi * fi;
      const float p3 = fr * fr * fr + fi * fi * fi;
      pr[j][0] = p1 * fr; pr[j][1] = p2 * fr; pr[j][2] = p3 * fr;
      pi[j][0] = p1 * fi; pi[j][1] = p2 * fi; pi[j][2] = p3 * fi;
    }
#pragma unroll
    for (int c = 0; c < 8; ++c)
#pragma unroll
      for (int p = 0; p < 2; ++p) {
        const float* Wr = w_coeff + ((c * 2 + p) * 2 + 0) * 21;
        const float* Wi = w_coeff + ((c * 2 + p) * 2 + 1) * 21;
        float s = 0.0f;
#pragma unroll
        for (int jj = 0; jj < 7; ++jj) {
          const int j2 = jj + (jj >= c);
#pragma unroll
          for (int d = 0; d < 3; ++d)
            s += pr[j2][d] * Wr[jj * 3 + d] + pi[j2][d] * Wi[jj * 3 + d];
        }
        const float wo = p ? w_out_imag[c * W_ + w] : w_out_real[c * W_ + w];
        atomicAdd(&O[g][c * 2 + p], s * wo);   // ds_add_f32
      }
  }
  __syncthreads();
  if (lane < 16) outbuf[b * 16 + lane] = O[g][lane];
}

// ---------------------------------------------------------------------------
// Kernel 3: batchnorm over axes (B, 2) per channel; wave32 shfl reductions.
// ---------------------------------------------------------------------------
__global__ __launch_bounds__(256) void bnorm(
    const float* __restrict__ outbuf,
    const float* __restrict__ gamma,
    const float* __restrict__ beta,
    float* __restrict__ out) {
  const int t = threadIdx.x, c = t >> 5, lane = t & 31;
  float vals[8];
  float s = 0.0f, s2 = 0.0f;
#pragma unroll
  for (int i = 0; i < 8; ++i) {
    const int bp = lane + 32 * i;          // 0..255 over (b,p)
    const int b = bp >> 1, p = bp & 1;
    const float v = outbuf[b * 16 + c * 2 + p];
    vals[i] = v; s += v; s2 += v * v;
  }
#pragma unroll
  for (int off = 16; off; off >>= 1) {
    s  += __shfl_xor(s,  off, 32);
    s2 += __shfl_xor(s2, off, 32);
  }
  const float mean = s * (1.0f / 256.0f);
  const float var  = s2 * (1.0f / 256.0f) - mean * mean;
  const float inv  = rsqrtf(var + EPS_) * gamma[c];
  const float bb   = beta[c];
#pragma unroll
  for (int i = 0; i < 8; ++i) {
    const int bp = lane + 32 * i;
    const int b = bp >> 1, p = bp & 1;
    out[(b * C_ + c) * 2 + p] = (vals[i] - mean) * inv + bb;
  }
}

// ---------------------------------------------------------------------------
extern "C" void kernel_launch(void* const* d_in, const int* in_sizes, int n_in,
                              void* d_out, int out_size, void* d_ws, size_t ws_size,
                              hipStream_t stream) {
  const float* x        = (const float*)d_in[0];
  const float* w_in_r   = (const float*)d_in[1];
  const float* w_in_i   = (const float*)d_in[2];
  const float* w_coeff  = (const float*)d_in[3];
  const float* w_out_r  = (const float*)d_in[4];
  const float* w_out_i  = (const float*)d_in[5];
  const float* bn_gamma = (const float*)d_in[6];
  const float* bn_beta  = (const float*)d_in[7];

  // workspace layout: [0,8MiB) Bfrag bf16 | [8,12MiB) Ypart f32 | [12MiB,..) outbuf
  unsigned short* Bfrag = (unsigned short*)d_ws;
  float* Ypart  = (float*)((char*)d_ws + (size_t)(8u << 20));
  float* outbuf = (float*)((char*)d_ws + (size_t)(12u << 20));

  prep_bfrag<<<1024, 256, 0, stream>>>(w_in_r, w_in_i, Bfrag);

  dim3 grid(8, KC_);
  fir_wmma<<<grid, 256, 0, stream>>>(x, Bfrag, Ypart);

  finish<<<B_ / 8, 256, 0, stream>>>(Ypart, w_coeff, w_out_r, w_out_i, outbuf);

  bnorm<<<1, 256, 0, stream>>>(outbuf, bn_gamma, bn_beta, (float*)d_out);
}
